// AbstractRNN_17652315587060
// MI455X (gfx1250) — compile-verified, matrix operand order checked by reading, not run on hardware
//
#include <hip/hip_runtime.h>
#include <hip/hip_bf16.h>
#include <math.h>

// ---------------------------------------------------------------------------
// Vanilla tanh RNN:  h_t = tanh(x_t @ W + h_{t-1} @ U + b), return h_{T-1}.
// B=64, T=1024, D=1024, H=1024.
//
// Phase 0: convert X -> f16; W,U -> transposed f16; zero h0 + barrier counter.
// Phase 1: Z = X@W + b (f16). WMMA f16 GEMM; per-WG 128x128 tile; X/W^T tiles
//          DMA'd into double-buffered LDS by the Tensor Data Mover
//          (tensor_load_to_lds + s_wait_tensorcnt), padded rows (40 halves)
//          for bank-conflict-free ds_load_b128 fragment reads.
// Phase 2: persistent 32-WG recurrence. Each WG keeps its 32 columns of U^T
//          resident in LDS (64.5 KB, padded stride 1032) for all 1024 steps;
//          h ping-pongs in global f16 (128 KB, L2-resident); one atomic
//          grid barrier per step.
// ---------------------------------------------------------------------------

typedef _Float16 half_t;
typedef __attribute__((ext_vector_type(16))) _Float16 v16h;
typedef __attribute__((ext_vector_type(8)))  float    v8f;
typedef __attribute__((ext_vector_type(4)))  unsigned tdm_v4u;
typedef __attribute__((ext_vector_type(8)))  int      tdm_v8i;
typedef __attribute__((ext_vector_type(4)))  int      tdm_v4i;

#define RNN_B 64
#define RNN_T 1024
#define RNN_D 1024
#define RNN_H 1024
#define P2_NWG 32       // persistent workgroups in phase 2
#define P1_LD  40       // phase-1 LDS tile row stride in halves (32 + 16B pad)
#define P2_LD  1032     // phase-2 LDS U^T row stride in halves (1024 + 16B pad)

#if __has_builtin(__builtin_amdgcn_tensor_load_to_lds)
#define HAVE_TDM 1
#else
#define HAVE_TDM 0
#endif

__device__ inline void tensor_wait0() {
#if __has_builtin(__builtin_amdgcn_s_wait_tensorcnt)
  __builtin_amdgcn_s_wait_tensorcnt(0);
#else
  asm volatile("s_wait_tensorcnt 0x0" ::: "memory");
#endif
}

// 2-D TDM load: tile (tile0 x tile1) of a (td0 x td1) f16 tensor, row stride
// stride0 elements, into LDS at byte offset lds_off, with optional LDS row
// padding (pad_interval/pad_amount codes per ISA D# group1).
__device__ inline void tdm_load_2d(unsigned lds_off, const void* gptr,
                                   unsigned td0, unsigned td1,
                                   unsigned tile0, unsigned tile1,
                                   unsigned stride0,
                                   unsigned pad_int, unsigned pad_amt,
                                   bool pad_en) {
#if HAVE_TDM
  unsigned long long ga = (unsigned long long)(uintptr_t)gptr;
  tdm_v4u g0;
  g0[0] = 1u;                                   // count=1, user descriptor
  g0[1] = lds_off;                              // lds_addr
  g0[2] = (unsigned)ga;                         // global_addr[31:0]
  g0[3] = (unsigned)((ga >> 32) & 0x01FFFFFFull) | (2u << 30);  // addr hi, type=2
  tdm_v8i g1;
  g1[0] = (int)((1u << 16) |                    // data_size = 2 bytes
                (pad_en ? (1u << 20) : 0u) |
                ((pad_int & 7u) << 22) | ((pad_amt & 127u) << 25));
  g1[1] = (int)((td0 & 0xFFFFu) << 16);                          // dim0 lo
  g1[2] = (int)(((td0 >> 16) & 0xFFFFu) | ((td1 & 0xFFFFu) << 16));
  g1[3] = (int)(((td1 >> 16) & 0xFFFFu) | ((tile0 & 0xFFFFu) << 16));
  g1[4] = (int)(tile1 & 0xFFFFu);                                // tile_dim1
  g1[5] = (int)stride0;                                          // dim0 stride
  g1[6] = 0;
  g1[7] = 0;
  tdm_v4i z4 = {0, 0, 0, 0};
#if __clang_major__ >= 23
  tdm_v8i z8 = {0, 0, 0, 0, 0, 0, 0, 0};
  __builtin_amdgcn_tensor_load_to_lds(g0, g1, z4, z4, z8, 0);
#else
  __builtin_amdgcn_tensor_load_to_lds(g0, g1, z4, z4, 0);
#endif
#endif
}

// WMMA 16-bit A-operand layout: lane = m + 16*hiK; halves 0..7 = K[hi*8..+7],
// halves 8..15 = K[16+hi*8..+7] -> two contiguous 16B runs per lane.
__device__ inline v16h load_frag(const half_t* __restrict__ base,
                                 int row0, int col0, int ld, int lane) {
  const int r  = lane & 15;
  const int hi = lane >> 4;
  const half_t* p = base + (size_t)(row0 + r) * (size_t)ld + col0 + hi * 8;
  union { v16h v; uint4 u[2]; } f;
  f.u[0] = *reinterpret_cast<const uint4*>(p);
  f.u[1] = *reinterpret_cast<const uint4*>(p + 16);
  return f.v;
}

// Same fragment pattern from an LDS tile with row stride ldh halves.
__device__ inline v16h lds_frag(const half_t* s, int row0, int col0, int ldh,
                                int lane) {
  const int r  = lane & 15;
  const int hi = lane >> 4;
  const half_t* p = s + (row0 + r) * ldh + col0 + hi * 8;
  union { v16h v; uint4 u[2]; } f;
  f.u[0] = *reinterpret_cast<const uint4*>(p);
  f.u[1] = *reinterpret_cast<const uint4*>(p + 16);
  return f.v;
}

__device__ inline v8f wmma_f16(v16h a, v16h b, v8f c) {
  return __builtin_amdgcn_wmma_f32_16x16x32_f16(
      false, a, false, b, (short)0, c, false, false);
}

// ------------------------------- Phase 0 -----------------------------------

__global__ void k_cvt_f16(const float* __restrict__ src, half_t* __restrict__ dst,
                          size_t n) {
  size_t i = (size_t)blockIdx.x * blockDim.x + threadIdx.x;
  if (i < n) dst[i] = (half_t)src[i];
}

__global__ void k_transpose_f16(const float* __restrict__ src, half_t* __restrict__ dst,
                                int rows, int cols) {
  int i = blockIdx.x * blockDim.x + threadIdx.x;
  if (i < rows * cols) {
    int r = i / cols, c = i % cols;
    dst[(size_t)c * rows + r] = (half_t)src[i];
  }
}

__global__ void k_zero_h16(half_t* __restrict__ p, int n) {
  int i = blockIdx.x * blockDim.x + threadIdx.x;
  if (i < n) p[i] = (half_t)0.0f;
}

__global__ void k_zero_u32(unsigned* __restrict__ p, int n) {
  int i = blockIdx.x * blockDim.x + threadIdx.x;
  if (i < n) p[i] = 0u;
}

// ------------------------------- Phase 1 -----------------------------------
// Grid: x = N/128 = 8, y = M/128 = 512. 256 threads = 8 waves.
// Wave (mq=w>>2, np=w&3): 64(M) x 32(N) = 4x2 WMMA tiles.
// Per k-step both 128x32 LDS tiles are filled by TDM (wave 0 issues, padded
// to 40-half rows), double-buffered against the WMMA compute.

__device__ inline void p1_stage(half_t* dstX, half_t* dstW,
                                const half_t* __restrict__ Xh,
                                const half_t* __restrict__ WT,
                                int m_blk, int n_blk, int k0, int tid) {
#if HAVE_TDM
  if ((tid >> 5) == 0) {
    // pad_interval code 3 = 16 dwords (64B row), pad_amount code 3 = 4 dwords
    tdm_load_2d((unsigned)(uintptr_t)dstX, Xh + (size_t)m_blk * RNN_D + k0,
                RNN_D, RNN_B * RNN_T, 32, 128, RNN_D, 3, 3, true);
    tdm_load_2d((unsigned)(uintptr_t)dstW, WT + (size_t)n_blk * RNN_D + k0,
                RNN_D, RNN_H, 32, 128, RNN_D, 3, 3, true);
  }
#else
  for (int c = tid; c < 512; c += 256) {
    const int row = c >> 2, cc = (c & 3) * 8;
    *reinterpret_cast<uint4*>(dstX + row * P1_LD + cc) =
        *reinterpret_cast<const uint4*>(Xh + (size_t)(m_blk + row) * RNN_D + k0 + cc);
    *reinterpret_cast<uint4*>(dstW + row * P1_LD + cc) =
        *reinterpret_cast<const uint4*>(WT + (size_t)(n_blk + row) * RNN_D + k0 + cc);
  }
#endif
}

__device__ inline void p1_stage_wait(int tid) {
#if HAVE_TDM
  if (tid < 32) tensor_wait0();
#endif
  __syncthreads();
}

__global__ __launch_bounds__(256)
void k_gemm_xw(const half_t* __restrict__ Xh,   // [B*T, D] f16
               const half_t* __restrict__ WT,   // [H, D]  f16 (W^T)
               const float*  __restrict__ bias, // [H]     f32
               half_t* __restrict__ Z) {        // [B*T, H] f16
  __shared__ half_t sX[2][128 * P1_LD];
  __shared__ half_t sW[2][128 * P1_LD];

  const int tid  = threadIdx.x;
  const int lane = tid & 31;
  const int wave = tid >> 5;
  const int m_blk = blockIdx.y * 128;
  const int n_blk = blockIdx.x * 128;
  const int mw = (wave >> 2) * 64;   // wave M offset inside the 128-row tile
  const int nw = (wave & 3) * 32;    // wave N offset inside the 128-col tile

  v8f acc[4][2] = {};

  p1_stage(sX[0], sW[0], Xh, WT, m_blk, n_blk, 0, tid);
  p1_stage_wait(tid);

  int cur = 0;
  for (int kb = 0; kb < RNN_D / 32; ++kb) {
    if (kb + 1 < RNN_D / 32)
      p1_stage(sX[cur ^ 1], sW[cur ^ 1], Xh, WT, m_blk, n_blk, (kb + 1) * 32, tid);

    const half_t* cx = sX[cur];
    const half_t* cw = sW[cur];
    v16h a[4], bt[2];
#pragma unroll
    for (int i = 0; i < 4; ++i) a[i] = lds_frag(cx, mw + 16 * i, 0, P1_LD, lane);
#pragma unroll
    for (int j = 0; j < 2; ++j) bt[j] = lds_frag(cw, nw + 16 * j, 0, P1_LD, lane);
#pragma unroll
    for (int i = 0; i < 4; ++i)
#pragma unroll
      for (int j = 0; j < 2; ++j)
        acc[i][j] = wmma_f16(a[i], bt[j], acc[i][j]);

    p1_stage_wait(tid);   // next tiles ready + this buffer fully consumed
    cur ^= 1;
  }

  // Epilogue: VGPR r, lanes 0..15 -> (m=r, n=lane); lanes 16..31 -> (m=r+8).
  const int nloc = lane & 15;
  const int hi   = lane >> 4;
#pragma unroll
  for (int i = 0; i < 4; ++i) {
#pragma unroll
    for (int j = 0; j < 2; ++j) {
      const int ncol = n_blk + nw + 16 * j + nloc;
      const float bv = bias[ncol];
#pragma unroll
      for (int r = 0; r < 8; ++r) {
        const int mrow = m_blk + mw + 16 * i + r + 8 * hi;
        Z[(size_t)mrow * RNN_H + ncol] = (half_t)(acc[i][j][r] + bv);
      }
    }
  }
}

// ------------------------------- Phase 2 -----------------------------------
// 32 persistent WGs x 256 threads; WG g owns columns [g*32, g*32+32).
// U^T slice (32 x 1024 f16) staged once into LDS with padded stride 1032
// (516 dwords = 4 mod 64 -> conflict-free b128 reads), reused all T steps.

__global__ __launch_bounds__(256)
void k_rnn(const half_t* __restrict__ Z,    // [B, T, H] f16
           const half_t* __restrict__ UT,   // [H, H] f16 (U^T)
           half_t* __restrict__ hb0,        // [B, H] f16 ping
           half_t* __restrict__ hb1,        // [B, H] f16 pong
           float*  __restrict__ out,        // [B, H] f32
           unsigned* __restrict__ cnt) {
  __shared__ half_t sU[32 * P2_LD];         // 66 KB

  const int lane = threadIdx.x & 31;
  const int wave = threadIdx.x >> 5;
  const int m0 = (wave >> 1) * 16;
  const int nw = (wave & 1) * 16;           // local N offset in [0,32)
  const int n0 = blockIdx.x * 32 + nw;
  const int nloc = lane & 15;
  const int hi   = lane >> 4;

  // Stage this WG's 32 rows of U^T into LDS (chunks of 8 halves).
  for (int c = threadIdx.x; c < 32 * 128; c += 256) {
    const int n  = c >> 7;
    const int kc = (c & 127) * 8;
    *reinterpret_cast<uint4*>(sU + n * P2_LD + kc) =
        *reinterpret_cast<const uint4*>(UT + (size_t)(blockIdx.x * 32 + n) * RNN_H + kc);
  }
  __syncthreads();

  for (int t = 0; t < RNN_T; ++t) {
    const half_t* hin  = (t & 1) ? hb1 : hb0;
    half_t*       hout = (t & 1) ? hb0 : hb1;

    v8f acc = {};
#pragma unroll 4
    for (int k0 = 0; k0 < RNN_H; k0 += 32) {
      v16h a = load_frag(hin, m0, k0, RNN_H, lane);     // global, L2-hot
      v16h b = lds_frag(sU, nw, k0, P2_LD, lane);       // LDS-resident U^T
      acc = wmma_f16(a, b, acc);
    }

    const int col = n0 + nloc;
#pragma unroll
    for (int r = 0; r < 8; ++r) {
      const int row = m0 + r + 8 * hi;  // batch index 0..63
      const float z = (float)Z[((size_t)row * RNN_T + t) * RNN_H + col];
      const float v = tanhf(acc[r] + z);
      hout[row * RNN_H + col] = (half_t)v;
      if (t == RNN_T - 1) out[row * RNN_H + col] = v;
    }

    // grid barrier: step t complete when counter reaches (t+1)*NWG
    __syncthreads();
    if (threadIdx.x == 0) {
      __hip_atomic_fetch_add(cnt, 1u, __ATOMIC_RELEASE, __HIP_MEMORY_SCOPE_AGENT);
      const unsigned target = (unsigned)(t + 1) * (unsigned)P2_NWG;
      while (__hip_atomic_load(cnt, __ATOMIC_ACQUIRE, __HIP_MEMORY_SCOPE_AGENT) < target)
        __builtin_amdgcn_s_sleep(1);
    }
    __syncthreads();
  }
}

// ------------------------------- Launch ------------------------------------

extern "C" void kernel_launch(void* const* d_in, const int* in_sizes, int n_in,
                              void* d_out, int out_size, void* d_ws, size_t ws_size,
                              hipStream_t stream) {
  const float* x = (const float*)d_in[0];  // [B, T, D]
  const float* W = (const float*)d_in[1];  // [D, H]
  const float* U = (const float*)d_in[2];  // [H, H]
  const float* b = (const float*)d_in[3];  // [H]
  float* out = (float*)d_out;              // [B, H]

  char* ws = (char*)d_ws;
  size_t off = 0;
  auto take = [&](size_t bytes) -> char* {
    char* p = ws + off;
    off = (off + bytes + 255) & ~(size_t)255;
    return p;
  };

  const size_t nX = (size_t)RNN_B * RNN_T * RNN_D;
  half_t*   Xh  = (half_t*)take(nX * sizeof(half_t));                 // 128 MB
  half_t*   Z   = (half_t*)take((size_t)RNN_B * RNN_T * RNN_H * 2);   // 128 MB
  half_t*   WT  = (half_t*)take((size_t)RNN_D * RNN_H * 2);           //   2 MB
  half_t*   UT  = (half_t*)take((size_t)RNN_H * RNN_H * 2);           //   2 MB
  half_t*   hb0 = (half_t*)take((size_t)RNN_B * RNN_H * 2);           // 128 KB
  half_t*   hb1 = (half_t*)take((size_t)RNN_B * RNN_H * 2);           // 128 KB
  unsigned* cnt = (unsigned*)take(256);

  // Phase 0
  k_zero_u32<<<1, 64, 0, stream>>>(cnt, 1);
  k_zero_h16<<<(RNN_B * RNN_H + 255) / 256, 256, 0, stream>>>(hb0, RNN_B * RNN_H);
  k_cvt_f16<<<(int)((nX + 255) / 256), 256, 0, stream>>>(x, Xh, nX);
  k_transpose_f16<<<(RNN_D * RNN_H + 255) / 256, 256, 0, stream>>>(W, WT, RNN_D, RNN_H);
  k_transpose_f16<<<(RNN_H * RNN_H + 255) / 256, 256, 0, stream>>>(U, UT, RNN_H, RNN_H);

  // Phase 1: Z = X@W + b
  dim3 g1(RNN_H / 128, (RNN_B * RNN_T) / 128);
  k_gemm_xw<<<g1, 256, 0, stream>>>(Xh, WT, b, Z);

  // Phase 2: persistent recurrence
  k_rnn<<<P2_NWG, 256, 0, stream>>>(Z, UT, hb0, hb1, out, cnt);
}
